// MPNNLayer_72043781423557
// MI455X (gfx1250) — matrix operand order repaired and checked by reading, not compiled, not run on hardware
//
#include <hip/hip_runtime.h>

// MI455X (gfx1250, wave32). Memory-bound problem: ~1.1 GB traffic @ 23.3 TB/s.
// All heavy GEMMs use v_wmma_f32_16x16x32_bf16 (fp32 accumulate) so the WMMA
// units keep the kernel on the bandwidth roof instead of the fp32 VALU roof.

typedef __attribute__((ext_vector_type(16))) __bf16 v16bf;
typedef __attribute__((ext_vector_type(8)))  float  v8f;

#define BB   4
#define NN   512
#define CC   128          // F == FE == FG == MID == OUT == 128
#define BIGN 1000000.0f

#define ROWB2 272         // LDS row stride (bytes) for 128 bf16 + pad, 16B multiple
#define ROWB3 528         // LDS row stride (bytes) for 256 bf16 + pad, 16B multiple

__device__ __forceinline__ unsigned short f2bf(float f) {
    unsigned int u = __float_as_uint(f);
    unsigned int r = u + 0x7FFFu + ((u >> 16) & 1u);   // round-to-nearest-even
    return (unsigned short)(r >> 16);
}
__device__ __forceinline__ unsigned int pk2bf(float a, float b) {
    return (unsigned int)f2bf(a) | ((unsigned int)f2bf(b) << 16);
}

union AFrag { v16bf v; uint4 q[2]; };

// ---------------------------------------------------------------------------
// Kernel 1: pack weights into WMMA B-fragment layout (bf16).
// B matrix 32x16 bf16 (K x N): lanes 0-15 hold col N=lane, K={0..7,16..23};
// lanes 16-31 hold col N=lane-16, K={8..15,24..31} (matches A fragment order).
// Fragment f stored as 512 bf16: element index = lane*16 + e.
// frags 0..31  : w_me   (ntile 0..7 x kstep 0..3)
// frags 32..95 : [w_o1 ; w_o2]  (ntile 0..7 x kstep 0..7, ks<4 -> w_o1)
// ---------------------------------------------------------------------------
__global__ __launch_bounds__(256) void mpnn_pack_weights(
    const float* __restrict__ w_me, const float* __restrict__ w_o1,
    const float* __restrict__ w_o2,
    unsigned short* __restrict__ wme_pk, unsigned short* __restrict__ wo_pk)
{
    int idx = blockIdx.x * 256 + threadIdx.x;
    if (idx >= 96 * 512) return;
    int f = idx >> 9;
    int r = idx & 511;
    int L = r >> 4;               // lane 0..31
    int e = r & 15;               // element 0..15
    int khalf = (L < 16) ? 0 : 8;
    int ncol  = L & 15;
    if (f < 32) {
        int nt = f >> 2, ks = f & 3;
        int k = ks * 32 + khalf + e + (e >= 8 ? 8 : 0);
        int n = nt * 16 + ncol;
        wme_pk[idx] = f2bf(w_me[k * CC + n]);
    } else {
        int f2 = f - 32;
        int nt = f2 >> 3, ks = f2 & 7;
        int k = (ks & 3) * 32 + khalf + e + (e >= 8 ? 8 : 0);
        int n = nt * 16 + ncol;
        const float* src = (ks < 4) ? w_o1 : w_o2;
        wo_pk[(size_t)f2 * 512 + r] = f2bf(src[k * CC + n]);
    }
}

// ---------------------------------------------------------------------------
// Kernel 2: small projections.
//   msg2[g,c]   = node[g]·w_m2[:,c] + b_m2[c]
//   base1g[g,c] = node[g]·w_m1[:,c] + b_m1[c] + graph[b]·w_mg[:,c] + b_mg[c]
// ---------------------------------------------------------------------------
__global__ __launch_bounds__(128) void mpnn_proj(
    const float* __restrict__ node, const float* __restrict__ graph,
    const float* __restrict__ w_m1, const float* __restrict__ b_m1,
    const float* __restrict__ w_m2, const float* __restrict__ b_m2,
    const float* __restrict__ w_mg, const float* __restrict__ b_mg,
    float* __restrict__ msg2, float* __restrict__ base1g)
{
    int g = blockIdx.x;                 // 0..2047  (b*512 + j)
    int c = threadIdx.x;                // 0..127
    int b = g >> 9;
    const float* nr = node + (size_t)g * CC;
    const float* gr = graph + (size_t)b * CC;
    float s2 = b_m2[c];
    float s1 = b_m1[c] + b_mg[c];
    #pragma unroll 4
    for (int k = 0; k < CC; ++k) {
        float nv = nr[k];
        s2 += nv * w_m2[k * CC + c];
        s1 += nv * w_m1[k * CC + c];
    }
    #pragma unroll 4
    for (int k = 0; k < CC; ++k)
        s1 += gr[k] * w_mg[k * CC + c];
    msg2[(size_t)g * CC + c]   = s2;
    base1g[(size_t)g * CC + c] = s1;
}

// ---------------------------------------------------------------------------
// Kernel 3: the big fused kernel.
//   For WG (b, jt, ng): stream i = 0..511.
//   msg_e[b,i, jt*16+jl, n] = edge[b,i,j,:] @ w_me + b_me   (bf16 WMMA)
//   redmax[jl,n] = max over i with adj[b,i,j] of (msg_e + msg2[b,i,n])
//   red[b,j,n]   = redmax + base1g[b,j,n]
// A-fragment layout per ISA 7.12.2 (16-bit A 16x32): lane<16 holds row M=lane,
// K in {klo..klo+7, klo+16..klo+23}, klo = 0 (lanes 0-15) / 8 (lanes 16-31)
// -> two contiguous 16B LDS reads per fragment.
// C/D layout: VGPR r, lane L -> M = r + 8*(L>>4), N = L&15.
// ---------------------------------------------------------------------------
__global__ __launch_bounds__(128) void mpnn_edge(
    const float* __restrict__ edge,            // [B,N,N,128]
    const unsigned char* __restrict__ adj,     // [B,N,N] bool
    const float* __restrict__ msg2,            // [B,N,128]
    const float* __restrict__ base1g,          // [B,N,128]
    const float* __restrict__ b_me,            // [128]
    const unsigned short* __restrict__ wme_pk, // 32 frags * 512 bf16
    float* __restrict__ out_msge,              // [B,N,N,128]
    float* __restrict__ red)                   // [B,N,128]
{
    __shared__ __align__(16) unsigned char ldsA[16 * ROWB2];

    int wg = blockIdx.x;            // 256 WGs: b(4) x jt(32) x ng(2)
    int ng = wg & 1;
    int jt = (wg >> 1) & 31;
    int b  = wg >> 6;

    int t    = threadIdx.x;         // 128 threads = 4 waves
    int wave = t >> 5;
    int lane = t & 31;
    int m    = lane & 15;
    int hi   = lane >> 4;
    int klo  = hi ? 8 : 0;
    int ntile = ng * 4 + wave;
    int n    = ntile * 16 + m;

    // Persistent B fragments for this wave's 16 output channels.
    v16bf bfrag[4];
    #pragma unroll
    for (int ks = 0; ks < 4; ++ks)
        bfrag[ks] = ((const v16bf*)(wme_pk + (size_t)(ntile * 4 + ks) * 512))[lane];

    float bias = b_me[n];
    float redacc[8];
    #pragma unroll
    for (int r = 0; r < 8; ++r) redacc[r] = -BIGN;

    // Cooperative loader role: row rr (0..15), 16-float chunk c8 (0..7).
    int rr = t >> 3;
    int c8 = t & 7;
    const float4* srcBase = (const float4*)(edge
        + (((size_t)(b * NN) * NN) + (size_t)(jt * 16 + rr) * CC) ) + c8 * 4;
    // stride between i values in float4 units: N*128/4
    const size_t istride4 = (size_t)NN * CC / 4;
    uint4* dst = (uint4*)(ldsA + rr * ROWB2 + c8 * 32);

    const unsigned char* aRow = ldsA + m * ROWB2;
    const unsigned char* adjBase = adj + ((size_t)b * NN) * NN + jt * 16 + hi * 8;
    const float* msg2Base = msg2 + (size_t)b * NN * CC + n;
    float* outBase = out_msge + (((size_t)(b * NN) * NN) + (size_t)(jt * 16)) * CC + n;

    for (int i = 0; i < NN; ++i) {
        // ---- cooperative fp32 -> bf16 tile load into LDS ----
        {
            const float4* s4 = srcBase + (size_t)i * istride4;
            float4 f0 = s4[0], f1 = s4[1], f2 = s4[2], f3 = s4[3];
            dst[0] = make_uint4(pk2bf(f0.x, f0.y), pk2bf(f0.z, f0.w),
                                pk2bf(f1.x, f1.y), pk2bf(f1.z, f1.w));
            dst[1] = make_uint4(pk2bf(f2.x, f2.y), pk2bf(f2.z, f2.w),
                                pk2bf(f3.x, f3.y), pk2bf(f3.z, f3.w));
            if (i + 1 < NN)
                __builtin_prefetch((const void*)(s4 + istride4), 0, 3);
        }
        __syncthreads();

        // ---- 16x16 tile: 4 chained K=32 bf16 WMMAs, seeded with b_me ----
        v8f acc;
        #pragma unroll
        for (int r = 0; r < 8; ++r) acc[r] = bias;
        #pragma unroll
        for (int ks = 0; ks < 4; ++ks) {
            AFrag af;
            const uint4* ap = (const uint4*)(aRow + (ks * 32 + klo) * 2);
            af.q[0] = ap[0];
            af.q[1] = ap[2];     // +32 bytes: K = klo+16 .. klo+23
            acc = __builtin_amdgcn_wmma_f32_16x16x32_bf16(
                false, af.v, false, bfrag[ks], (short)0, acc, false, false);
        }
        __syncthreads();

        // ---- store msg_e, fused masked max with msg2 ----
        unsigned long long av =
            *(const unsigned long long*)(adjBase + (size_t)i * NN);
        float m2v = msg2Base[(size_t)i * CC];
        float* orow = outBase + (size_t)i * NN * CC;
        #pragma unroll
        for (int r = 0; r < 8; ++r) {
            int jl = r + 8 * hi;
            float v = acc[r];
            orow[(size_t)jl * CC] = v;
            if ((av >> (8 * r)) & 0xFF) {
                float cand = v + m2v;
                redacc[r] = fmaxf(redacc[r], cand);
            }
        }
    }

    // ---- epilogue: red = redmax + (msg1 + msgg + biases) ----
    #pragma unroll
    for (int r = 0; r < 8; ++r) {
        int j = jt * 16 + r + 8 * hi;
        size_t idx = ((size_t)b * NN + j) * CC + n;
        red[idx] = redacc[r] + base1g[idx];
    }
}

// ---------------------------------------------------------------------------
// Kernel 4: ret = relu(node @ w_o1 + red @ w_o2 + b_o1 + b_o2)
// One K=256 WMMA GEMM over the concatenated [node | red] A-tile.
// ---------------------------------------------------------------------------
__global__ __launch_bounds__(128) void mpnn_out(
    const float* __restrict__ node, const float* __restrict__ red,
    const float* __restrict__ b_o1, const float* __restrict__ b_o2,
    const unsigned short* __restrict__ wo_pk,   // 64 frags * 512 bf16
    float* __restrict__ out_ret)                // [B,N,128]
{
    __shared__ __align__(16) unsigned char ldsA[16 * ROWB3];

    int wg = blockIdx.x;        // 256 WGs: rowtile(128) x ng(2)
    int ng = wg & 1;
    int rt = wg >> 1;

    int t    = threadIdx.x;
    int wave = t >> 5;
    int lane = t & 31;
    int m    = lane & 15;
    int hi   = lane >> 4;
    int klo  = hi ? 8 : 0;
    int ntile = ng * 4 + wave;
    int n    = ntile * 16 + m;

    // cooperative load: 16 rows x 256 bf16  (k 0..127 = node, 128..255 = red)
    {
        int rr = t >> 3;
        int c8 = t & 7;                  // chunk of 32 floats
        int g  = rt * 16 + rr;
        const float* src = (c8 < 4)
            ? (node + (size_t)g * CC + c8 * 32)
            : (red  + (size_t)g * CC + (c8 - 4) * 32);
        const float4* s4 = (const float4*)src;
        uint4* dst = (uint4*)(ldsA + rr * ROWB3 + c8 * 64);
        #pragma unroll
        for (int q = 0; q < 4; ++q) {
            float4 fa = s4[2 * q], fb = s4[2 * q + 1];
            dst[q] = make_uint4(pk2bf(fa.x, fa.y), pk2bf(fa.z, fa.w),
                                pk2bf(fb.x, fb.y), pk2bf(fb.z, fb.w));
        }
    }
    __syncthreads();

    float bias = b_o1[n] + b_o2[n];
    v8f acc;
    #pragma unroll
    for (int r = 0; r < 8; ++r) acc[r] = bias;

    const unsigned char* aRow = ldsA + m * ROWB3;
    #pragma unroll
    for (int ks = 0; ks < 8; ++ks) {
        AFrag af;
        const uint4* ap = (const uint4*)(aRow + (ks * 32 + klo) * 2);
        af.q[0] = ap[0];
        af.q[1] = ap[2];
        v16bf bf = ((const v16bf*)(wo_pk + (size_t)(ntile * 8 + ks) * 512))[lane];
        acc = __builtin_amdgcn_wmma_f32_16x16x32_bf16(
            false, af.v, false, bf, (short)0, acc, false, false);
    }

    #pragma unroll
    for (int r = 0; r < 8; ++r) {
        int g = rt * 16 + r + 8 * hi;
        out_ret[(size_t)g * CC + n] = fmaxf(acc[r], 0.0f);
    }
}

// ---------------------------------------------------------------------------
extern "C" void kernel_launch(void* const* d_in, const int* in_sizes, int n_in,
                              void* d_out, int out_size, void* d_ws, size_t ws_size,
                              hipStream_t stream) {
    (void)in_sizes; (void)n_in; (void)out_size; (void)ws_size;
    const float* node_fts  = (const float*)d_in[0];
    const float* edge_fts  = (const float*)d_in[1];
    const float* graph_fts = (const float*)d_in[2];
    const unsigned char* adj_mat = (const unsigned char*)d_in[3];   // bool
    // d_in[4] = hidden (unused by reference)
    const float* w_m1 = (const float*)d_in[5];
    const float* b_m1 = (const float*)d_in[6];
    const float* w_m2 = (const float*)d_in[7];
    const float* b_m2 = (const float*)d_in[8];
    const float* w_me = (const float*)d_in[9];
    const float* b_me = (const float*)d_in[10];
    const float* w_mg = (const float*)d_in[11];
    const float* b_mg = (const float*)d_in[12];
    const float* w_o1 = (const float*)d_in[13];
    const float* b_o1 = (const float*)d_in[14];
    const float* w_o2 = (const float*)d_in[15];
    const float* b_o2 = (const float*)d_in[16];

    float* out_ret  = (float*)d_out;                       // [4,512,128]
    float* out_msge = (float*)d_out + (size_t)BB * NN * CC; // [4,512,512,128]

    unsigned char* ws = (unsigned char*)d_ws;
    unsigned short* wme_pk = (unsigned short*)ws;                        // 32 KB
    unsigned short* wo_pk  = (unsigned short*)(ws + 32768);              // 64 KB
    float* msg2   = (float*)(ws + 98304);                                // 1 MB
    float* base1g = (float*)(ws + 98304 + 1048576);                      // 1 MB
    float* redws  = (float*)(ws + 98304 + 2 * 1048576);                  // 1 MB

    mpnn_pack_weights<<<192, 256, 0, stream>>>(w_me, w_o1, w_o2, wme_pk, wo_pk);
    mpnn_proj<<<BB * NN, 128, 0, stream>>>(node_fts, graph_fts,
                                           w_m1, b_m1, w_m2, b_m2, w_mg, b_mg,
                                           msg2, base1g);
    mpnn_edge<<<256, 128, 0, stream>>>(edge_fts, adj_mat, msg2, base1g, b_me,
                                       wme_pk, out_msge, redws);
    mpnn_out<<<256, 128, 0, stream>>>(node_fts, redws, b_o1, b_o2, wo_pk, out_ret);
}
